// SelfAttention_83597243449669
// MI455X (gfx1250) — compile-verified, matrix operand order checked by reading, not run on hardware
//
#include <hip/hip_runtime.h>
#include <math.h>

// ---------------------------------------------------------------------------
// Causal self-attention, B=4 T=4096 D=512, fp32 in/out.
// bf16 WMMA (v_wmma_f32_16x16x32_bf16) with f32 accumulate:
//   1) convert x, Wqkv, Wout to bf16 in workspace
//   2) GEMM1: qkv = x @ Wqkv + bqkv  -> bf16 Q[b,t,d], K[b,t,d], V^T[b,d,t]
//   3) flash-attention (online softmax, causal), never materializes T x T
//   4) GEMM2: out = attn @ Wout + bout (f32 output)
//
// WMMA layout assumptions (CDNA5 ISA, 16-bit A 16x32):
//   A frag: lane L holds row M=L%16; elements j<8 -> K=(L/16)*8+j,
//           j>=8 -> K=16+(L/16)*8+(j-8). Two contiguous 16B loads.
//   B frag: mirrored (lane = output column N, elements = K runs) — LDS tiles
//   are arranged so B fragment loads are also contiguous 16B loads.
//   C/D:    element e -> row M=e+8*(lane/16), col N=lane%16.
//
// Round-2 change: flash kernel no longer keeps Q fragments resident in VGPRs
// (was 128 VGPRs -> spills). Q is staged once into per-wave LDS and fragments
// are reloaded in the d-loop; register budget drops to ~350 VGPRs (no spill).
// ---------------------------------------------------------------------------

typedef __attribute__((ext_vector_type(8)))  float  v8f;
typedef __attribute__((ext_vector_type(8)))  __bf16 v8bf;
typedef __attribute__((ext_vector_type(16))) __bf16 v16bf;

__device__ __forceinline__ v16bf cat8(v8bf lo, v8bf hi) {
  return __builtin_shufflevector(lo, hi, 0, 1, 2, 3, 4, 5, 6, 7,
                                 8, 9, 10, 11, 12, 13, 14, 15);
}
__device__ __forceinline__ v16bf load_frag(const __bf16* p0, const __bf16* p1) {
  return cat8(*(const v8bf*)p0, *(const v8bf*)p1);
}
__device__ __forceinline__ v8f wmma_bf16(v16bf a, v16bf b, v8f c) {
  return __builtin_amdgcn_wmma_f32_16x16x32_bf16(
      /*neg_a=*/false, a, /*neg_b=*/false, b,
      /*c_mod=*/(short)0, c, /*reuse_a=*/false, /*reuse_b=*/false);
}

// ---------------------------------------------------------------------------
__global__ void f32_to_bf16_kernel(const float* __restrict__ in,
                                   __bf16* __restrict__ out, int n) {
  int i = blockIdx.x * blockDim.x + threadIdx.x;
  int stride = gridDim.x * blockDim.x;
  for (; i < n; i += stride) out[i] = (__bf16)in[i];
}

// ---------------------------------------------------------------------------
// Tiled bf16 GEMM: C[M,N] = A[M,K] @ B[K,N] + bias.
// Block: 256 threads (8 waves), tile 128x64, BK=32. Wave -> 32x32 (2x2 WMMA).
// MODE 0: write f32 C row-major.  MODE 1: split qkv into Q / K / V^T (bf16).
template <int MODE>
__global__ __launch_bounds__(256) void gemm_bf16_kernel(
    const __bf16* __restrict__ A, const __bf16* __restrict__ Bm,
    const float* __restrict__ bias, float* __restrict__ Cf,
    __bf16* __restrict__ qo, __bf16* __restrict__ ko,
    __bf16* __restrict__ vto, int M, int N, int K) {
  __shared__ __align__(16) __bf16 Asl[128][32];  // row-major [m][k]
  __shared__ __align__(16) __bf16 Bsl[64][32];   // transposed  [n][k]

  const int tid = threadIdx.x;
  const int lane = tid & 31, wave = tid >> 5;
  const int khalf = lane >> 4, ncol = lane & 15;
  const int m0 = blockIdx.y * 128;
  const int n0 = blockIdx.x * 64;
  const int wm = wave & 3, wn = wave >> 2;

  const v8f vzero = {0.f, 0.f, 0.f, 0.f, 0.f, 0.f, 0.f, 0.f};
  v8f acc[2][2];
#pragma unroll
  for (int mi = 0; mi < 2; ++mi)
#pragma unroll
    for (int ni = 0; ni < 2; ++ni) acc[mi][ni] = vzero;

  for (int kt = 0; kt < K; kt += 32) {
    // Stage A tile: 128x32 bf16 = 512 chunks of 16B, 2 per thread.
#pragma unroll
    for (int i = 0; i < 2; ++i) {
      int c = tid * 2 + i;            // 0..511
      int r = c >> 2;                 // 4 chunks per row
      int c8 = (c & 3) * 8;
      *(v8bf*)&Asl[r][c8] =
          *(const v8bf*)(A + (size_t)(m0 + r) * K + kt + c8);
    }
    // Stage B tile transposed: read 16B rows along N, scatter to [n][k].
    {
      int r = tid >> 3;               // k row 0..31
      int cc = (tid & 7) * 8;         // n within tile
      v8bf t = *(const v8bf*)(Bm + (size_t)(kt + r) * N + n0 + cc);
#pragma unroll
      for (int j = 0; j < 8; ++j) Bsl[cc + j][r] = t[j];
    }
    __syncthreads();

    v16bf af[2], bf[2];
#pragma unroll
    for (int mi = 0; mi < 2; ++mi) {
      int r = 32 * wm + 16 * mi + ncol;
      af[mi] = load_frag(&Asl[r][khalf * 8], &Asl[r][16 + khalf * 8]);
    }
#pragma unroll
    for (int ni = 0; ni < 2; ++ni) {
      int cn = 32 * wn + 16 * ni + ncol;
      bf[ni] = load_frag(&Bsl[cn][khalf * 8], &Bsl[cn][16 + khalf * 8]);
    }
#pragma unroll
    for (int mi = 0; mi < 2; ++mi)
#pragma unroll
      for (int ni = 0; ni < 2; ++ni)
        acc[mi][ni] = wmma_bf16(af[mi], bf[ni], acc[mi][ni]);
    __syncthreads();
  }

  // Epilogue.
#pragma unroll
  for (int mi = 0; mi < 2; ++mi)
#pragma unroll
    for (int ni = 0; ni < 2; ++ni) {
      const int gn = n0 + 32 * wn + 16 * ni + ncol;
      const float bv = bias[gn];
#pragma unroll
      for (int e = 0; e < 8; ++e) {
        const int gm = m0 + 32 * wm + 16 * mi + khalf * 8 + e;
        const float v = acc[mi][ni][e] + bv;
        if (MODE == 0) {
          Cf[(size_t)gm * N + gn] = v;
        } else {
          const int bidx = gm >> 12;          // token row -> (batch, t)
          const int t = gm & 4095;
          const size_t tok = (size_t)gm;
          if (gn < 512)
            qo[tok * 512 + gn] = (__bf16)v;
          else if (gn < 1024)
            ko[tok * 512 + (gn - 512)] = (__bf16)v;
          else  // V stored transposed: vT[b][d][t]
            vto[((size_t)bidx * 512 + (gn - 1024)) * 4096 + t] = (__bf16)v;
        }
      }
    }
}

// ---------------------------------------------------------------------------
// Flash attention, causal. Grid: (T/128, B). Block: 256 threads (8 waves).
// Wave w owns query rows [bx*128 + 16w, +16). Output accumulator (16x512 f32,
// 256 VGPRs) lives in registers; Q (per-wave 16x512), K (32x512) and
// V^T (512x32) tiles live in LDS (200 KB dynamic, < 320 KB WGP budget).
__global__ __launch_bounds__(256) void flash_attn_kernel(
    const __bf16* __restrict__ qb, const __bf16* __restrict__ kb,
    const __bf16* __restrict__ vtb, __bf16* __restrict__ ob) {
  extern __shared__ __align__(16) char smem[];
  __bf16* Kl = (__bf16*)smem;        // [32][512] row-major (key, d)
  __bf16* Vl = Kl + 32 * 512;        // [512][32] (d, key)  == V^T tile
  __bf16* Pl = Vl + 512 * 32;        // [8 waves][16][32] P staging
  __bf16* Ql = Pl + 8 * 16 * 32;     // [8 waves][16][512] Q staging

  const int tid = threadIdx.x;
  const int lane = tid & 31, wave = tid >> 5;
  const int khalf = lane >> 4, ncol = lane & 15;
  const int b = blockIdx.y;
  const int qrow0 = blockIdx.x * 128 + wave * 16;
  const size_t bq = (size_t)b * 4096;
  const float scale = 0.04419417382415922f;  // 1/sqrt(512)

  // Stage this wave's 16x512 Q tile into its private LDS region. Only this
  // wave reads it back, and same-wave LDS ops are in-order -> no barrier.
  __bf16* Qw = Ql + wave * (16 * 512);
#pragma unroll
  for (int i = 0; i < 32; ++i) {
    int c = lane + i * 32;           // 0..1023 chunks of 16B
    int r = c >> 6, c8 = (c & 63) * 8;
    *(v8bf*)&Qw[r * 512 + c8] =
        *(const v8bf*)(qb + (bq + qrow0 + r) * 512 + c8);
  }

  const v8f vzero = {0.f, 0.f, 0.f, 0.f, 0.f, 0.f, 0.f, 0.f};
  v8f acc[32];
#pragma unroll
  for (int dt = 0; dt < 32; ++dt) acc[dt] = vzero;
  float mrun[8], lrun[8];
#pragma unroll
  for (int e = 0; e < 8; ++e) { mrun[e] = -1e30f; lrun[e] = 0.f; }

  __bf16* Pw = Pl + wave * (16 * 32);
  const int qrow_lane = wave * 16 + ncol;   // this lane's Q row in LDS
  const int jend = (blockIdx.x + 1) * 128;  // causal: keys < block row end

  for (int j0 = 0; j0 < jend; j0 += 32) {
    // Cooperative stage of K tile (32 keys x 512) row-major.
#pragma unroll
    for (int i = 0; i < 8; ++i) {
      int c = tid + i * 256;          // 0..2047 chunks of 16B
      int r = c >> 6, c8 = (c & 63) * 8;
      *(v8bf*)&Kl[r * 512 + c8] =
          *(const v8bf*)(kb + (bq + j0 + r) * 512 + c8);
    }
    // Cooperative stage of V^T tile (512 d x 32 keys).
#pragma unroll
    for (int i = 0; i < 8; ++i) {
      int c = tid + i * 256;
      int d = c >> 2, c8 = (c & 3) * 8;
      *(v8bf*)&Vl[d * 32 + c8] =
          *(const v8bf*)(vtb + ((size_t)b * 512 + d) * 4096 + j0 + c8);
    }
    if (j0 + 32 < jend) {  // hint next tiles (global_prefetch_b8)
      __builtin_prefetch(
          kb + (bq + j0 + 32 + (tid >> 3)) * 512 + (tid & 7) * 64, 0, 1);
      __builtin_prefetch(
          vtb + ((size_t)b * 512 + tid * 2) * 4096 + j0 + 32, 0, 1);
    }
    __syncthreads();

    if (j0 <= qrow0 + 15) {  // wave-uniform causal skip
      // S = Q K^T for two 16-key sub-tiles; Q fragments reloaded from LDS.
      v8f s0 = vzero, s1 = vzero;
#pragma unroll
      for (int d = 0; d < 16; ++d) {
        v16bf qf = load_frag(&Qw[ncol * 512 + d * 32 + khalf * 8],
                             &Qw[ncol * 512 + d * 32 + 16 + khalf * 8]);
        v16bf b0 = load_frag(&Kl[ncol * 512 + d * 32 + khalf * 8],
                             &Kl[ncol * 512 + d * 32 + 16 + khalf * 8]);
        v16bf b1 = load_frag(&Kl[(16 + ncol) * 512 + d * 32 + khalf * 8],
                             &Kl[(16 + ncol) * 512 + d * 32 + 16 + khalf * 8]);
        s0 = wmma_bf16(qf, b0, s0);
        s1 = wmma_bf16(qf, b1, s1);
      }
      // Online softmax (rows live at fixed element index across a 16-lane
      // half; xor-shuffles with offsets <16 stay inside the half).
      float p0[8], p1[8], corr[8];
#pragma unroll
      for (int e = 0; e < 8; ++e) {
        const int qr = qrow0 + khalf * 8 + e;
        float a0 = s0[e] * scale; if (j0 + ncol > qr)      a0 = -1e30f;
        float a1 = s1[e] * scale; if (j0 + 16 + ncol > qr) a1 = -1e30f;
        float tm = fmaxf(a0, a1);
        tm = fmaxf(tm, __shfl_xor(tm, 1, 32));
        tm = fmaxf(tm, __shfl_xor(tm, 2, 32));
        tm = fmaxf(tm, __shfl_xor(tm, 4, 32));
        tm = fmaxf(tm, __shfl_xor(tm, 8, 32));
        const float mnew = fmaxf(mrun[e], tm);
        corr[e] = __expf(mrun[e] - mnew);
        mrun[e] = mnew;
        const float e0 = __expf(a0 - mnew);
        const float e1 = __expf(a1 - mnew);
        float rs = e0 + e1;
        rs += __shfl_xor(rs, 1, 32);
        rs += __shfl_xor(rs, 2, 32);
        rs += __shfl_xor(rs, 4, 32);
        rs += __shfl_xor(rs, 8, 32);
        lrun[e] = lrun[e] * corr[e] + rs;
        p0[e] = e0; p1[e] = e1;
      }
#pragma unroll
      for (int dt = 0; dt < 32; ++dt)
#pragma unroll
        for (int e = 0; e < 8; ++e) acc[dt][e] *= corr[e];

      // Re-layout P (C layout -> A fragment) through per-wave LDS scratch.
#pragma unroll
      for (int e = 0; e < 8; ++e) {
        const int pr = khalf * 8 + e;
        Pw[pr * 32 + ncol]      = (__bf16)p0[e];
        Pw[pr * 32 + 16 + ncol] = (__bf16)p1[e];
      }
      v16bf pf = load_frag(&Pw[ncol * 32 + khalf * 8],
                           &Pw[ncol * 32 + 16 + khalf * 8]);
      // O += P V : 32 output d-tiles, one WMMA each (K = 32 keys).
#pragma unroll
      for (int dt = 0; dt < 32; ++dt) {
        v16bf vf = load_frag(&Vl[(dt * 16 + ncol) * 32 + khalf * 8],
                             &Vl[(dt * 16 + ncol) * 32 + 16 + khalf * 8]);
        acc[dt] = wmma_bf16(pf, vf, acc[dt]);
      }
    }
    __syncthreads();
  }

  // Normalize and write bf16 attention output [b,t,d].
  float rl[8];
#pragma unroll
  for (int e = 0; e < 8; ++e) rl[e] = 1.0f / lrun[e];
#pragma unroll
  for (int dt = 0; dt < 32; ++dt)
#pragma unroll
    for (int e = 0; e < 8; ++e)
      ob[(bq + qrow0 + khalf * 8 + e) * 512 + dt * 16 + ncol] =
          (__bf16)(acc[dt][e] * rl[e]);
}

// ---------------------------------------------------------------------------
extern "C" void kernel_launch(void* const* d_in, const int* in_sizes, int n_in,
                              void* d_out, int out_size, void* d_ws,
                              size_t ws_size, hipStream_t stream) {
  const float* x    = (const float*)d_in[0];  // [4,4096,512]
  const float* Wqkv = (const float*)d_in[1];  // [512,1536]
  const float* bqkv = (const float*)d_in[2];  // [1536]
  const float* Wout = (const float*)d_in[3];  // [512,512]
  const float* bout = (const float*)d_in[4];  // [512]
  float* out = (float*)d_out;                 // [4,4096,512] f32

  char* w = (char*)d_ws;
  auto take = [&](size_t bytes) {
    char* p = w;
    w += (bytes + 255) & ~(size_t)255;
    return p;
  };
  __bf16* xb  = (__bf16*)take((size_t)16384 * 512 * 2);
  __bf16* wqb = (__bf16*)take((size_t)512 * 1536 * 2);
  __bf16* wob = (__bf16*)take((size_t)512 * 512 * 2);
  __bf16* qb  = (__bf16*)take((size_t)16384 * 512 * 2);
  __bf16* kb  = (__bf16*)take((size_t)16384 * 512 * 2);
  __bf16* vtb = (__bf16*)take((size_t)16384 * 512 * 2);
  __bf16* ab  = (__bf16*)take((size_t)16384 * 512 * 2);

  f32_to_bf16_kernel<<<2048, 256, 0, stream>>>(x, xb, 16384 * 512);
  f32_to_bf16_kernel<<<512, 256, 0, stream>>>(Wqkv, wqb, 512 * 1536);
  f32_to_bf16_kernel<<<256, 256, 0, stream>>>(Wout, wob, 512 * 512);

  dim3 g1(1536 / 64, 16384 / 128);  // QKV projection -> Q, K, V^T (bf16)
  gemm_bf16_kernel<1><<<g1, 256, 0, stream>>>(xb, wqb, bqkv, nullptr, qb, kb,
                                              vtb, 16384, 1536, 512);

  // Dynamic LDS: (32*512 + 512*32 + 8*16*32 + 8*16*512) bf16 = 204800 B.
  dim3 g2(4096 / 128, 4);           // flash attention
  flash_attn_kernel<<<g2, 256, 204800, stream>>>(qb, kb, vtb, ab);

  dim3 g3(512 / 64, 16384 / 128);   // output projection (f32 out)
  gemm_bf16_kernel<0><<<g3, 256, 0, stream>>>(ab, wob, bout, out, nullptr,
                                              nullptr, nullptr, 16384, 512, 512);
}
